// expandVisAttn3_7_23356032156191
// MI455X (gfx1250) — matrix operand order repaired and checked by reading, not compiled, not run on hardware
//
#include <hip/hip_runtime.h>

typedef __bf16 bf16_t;
typedef __attribute__((ext_vector_type(8)))  __bf16 v8bf;
typedef __attribute__((ext_vector_type(16))) __bf16 v16bf;
typedef __attribute__((ext_vector_type(8)))  float  v8f;

#define DEVINL __device__ __forceinline__

// ---------------- geometry ----------------
constexpr int BATCH = 16384;
constexpr int CCH   = 256;   // C
constexpr int HDIM  = 512;   // H
constexpr int NHEAD = 8;
constexpr int HD    = 64;    // head dim
constexpr int ODIM  = 256;   // O
constexpr int BT    = 32;    // batch rows per block (two WMMA row tiles)

// ring positions (row-major nonzero order per stage), pos = r*7 + c
__constant__ int POS[49] = {
  24,
  16,17,18,23,25,30,31,32,
  8,9,10,11,12,15,19,22,26,29,33,36,37,38,39,40,
  0,1,2,3,4,5,6,7,13,14,20,21,27,28,34,35,41,42,43,44,45,46,47,48
};
__constant__ int SOFF[4] = {0, 1, 9, 25};
__constant__ int SM[4]   = {1, 8, 16, 24};

// ---------------- workspace layout (bf16 elements) ----------------
constexpr size_t OFF_Q1T   = 0;                        // [512][256]
constexpr size_t OFF_Q2T   = OFF_Q1T   + 131072;       // [512][512]
constexpr size_t OFF_WQ2DT = OFF_Q2T   + 262144;       // 4x [512][512]
constexpr size_t OFF_WK2D  = OFF_WQ2DT + 4*262144;     // 4x [256][512] (untransposed)
constexpr size_t OFF_WV2DT = OFF_WK2D  + 4*131072;     // 4x [512][256]
constexpr size_t OFF_WO2DT = OFF_WV2DT + 4*131072;     // 4x [512][512]
constexpr size_t OFF_WP2DT = OFF_WO2DT + 4*262144;     // 4x [64][512] (padded 49->64)
constexpr size_t OFF_WQ1DT = OFF_WP2DT + 4*32768;      // 3x [512][512]
constexpr size_t OFF_WK1D  = OFF_WQ1DT + 3*262144;     // 3x [512][512] (untransposed)
constexpr size_t OFF_WV1DT = OFF_WK1D  + 3*262144;     // 3x [512][512]
constexpr size_t OFF_WO1DT = OFF_WV1DT + 3*262144;     // 3x [512][512]
constexpr size_t OFF_OW1T  = OFF_WO1DT + 3*262144;     // [256][512]
constexpr size_t OFF_OW2T  = OFF_OW1T  + 131072;       // [256][256]
constexpr size_t OFF_REL   = OFF_OW2T  + 65536;        // [49][512]
constexpr size_t OFF_SA    = OFF_REL   + 25088;        // [B][4][512]
// total = OFF_SA + 16384*4*512 elements  (~81 MB bf16)

DEVINL v8f vzero() { v8f z = {0.f,0.f,0.f,0.f,0.f,0.f,0.f,0.f}; return z; }

// ---------------- WMMA fragment helpers ----------------
// A (16x32 bf16): lane<16 -> M=lane, K {0..7,16..23}; lane>=16 -> M=lane-16, K {8..15,24..31}
DEVINL v16bf ldfragA(const bf16_t* base, int stride) {
  int lane = threadIdx.x & 31;
  const bf16_t* q = base + (lane & 15) * stride + ((lane & 16) ? 8 : 0);
  union { struct { v8bf lo, hi; } s; v16bf v; } u;
  u.s.lo = *(const v8bf*)(q);
  u.s.hi = *(const v8bf*)(q + 16);
  return u.v;
}
// B (32x16 bf16): lane<16 -> N=lane, K {0..7,16..23}; lane>=16 -> N=lane-16, K {8..15,24..31}
DEVINL v16bf ldfragB(const bf16_t* base, int stride) {
  int lane = threadIdx.x & 31;
  const bf16_t* q = base + (lane & 15) * stride + ((lane & 16) ? 8 : 0);
  union { struct { v8bf lo, hi; } s; v16bf v; } u;
  u.s.lo = *(const v8bf*)(q);
  u.s.hi = *(const v8bf*)(q + 16);
  return u.v;
}
DEVINL v8f wmma_bf16(v16bf a, v16bf b, v8f c) {
  return __builtin_amdgcn_wmma_f32_16x16x32_bf16(false, a, false, b, (short)0, c, false, false);
}
// C/D (16x16 f32): lane<16 -> N=lane, M=vgpr; lane>=16 -> N=lane-16, M=8+vgpr
DEVINL void stfrag_bf16(bf16_t* cbase, int stride, v8f acc, bool relu) {
  int lane = threadIdx.x & 31;
  int n = lane & 15;
  int mb = (lane & 16) ? 8 : 0;
#pragma unroll
  for (int i = 0; i < 8; ++i) {
    float v = acc[i];
    if (relu) v = fmaxf(v, 0.f);
    cbase[(mb + i) * stride + n] = (bf16_t)v;
  }
}

// ---------------- generic GEMMs (M = 32 batch rows, two 16-row tiles share each B fragment) ----------------
// C[32][N] (LDS bf16) = A[32][K] (LDS bf16) x W, W pre-laid as [N][K] bf16 (ldw = K stride)
__device__ void gemm16(const bf16_t* A, int lda, const bf16_t* W, int ldw,
                       bf16_t* Cst, int ldc, int N, int K, bool relu) {
  int wave = threadIdx.x >> 5;
  for (int nt = wave; nt < (N >> 4); nt += 8) {
    v8f acc0 = vzero(), acc1 = vzero();
    const bf16_t* wp = W + (size_t)(nt << 4) * ldw;
    for (int k0 = 0; k0 < K; k0 += 32) {
      v16bf b = ldfragB(wp + k0, ldw);              // one B fragment ...
      acc0 = wmma_bf16(ldfragA(A + k0, lda), b, acc0);               // ... feeds rows 0-15
      acc1 = wmma_bf16(ldfragA(A + 16 * lda + k0, lda), b, acc1);    // ... and rows 16-31
    }
    stfrag_bf16(Cst + (nt << 4), ldc, acc0, relu);
    stfrag_bf16(Cst + 16 * ldc + (nt << 4), ldc, acc1, relu);
  }
}

// final GEMM writing f32 to global
__device__ void gemm16_out(const bf16_t* A, int lda, const bf16_t* W, int ldw,
                           float* outg, int ldo, int N, int K) {
  int wave = threadIdx.x >> 5;
  int lane = threadIdx.x & 31;
  int n = lane & 15, mb = (lane & 16) ? 8 : 0;
  for (int nt = wave; nt < (N >> 4); nt += 8) {
    v8f acc0 = vzero(), acc1 = vzero();
    const bf16_t* wp = W + (size_t)(nt << 4) * ldw;
    for (int k0 = 0; k0 < K; k0 += 32) {
      v16bf b = ldfragB(wp + k0, ldw);
      acc0 = wmma_bf16(ldfragA(A + k0, lda), b, acc0);
      acc1 = wmma_bf16(ldfragA(A + 16 * lda + k0, lda), b, acc1);
    }
#pragma unroll
    for (int i = 0; i < 8; ++i) {
      outg[(size_t)(mb + i) * ldo + (nt << 4) + n]      = acc0[i];
      outg[(size_t)(mb + i + 16) * ldo + (nt << 4) + n] = acc1[i];
    }
  }
}

// folded-key slice: for all heads h, C[b][h][ec] = sum_d qh[b][h*64+d] * W[(nBase+ec)*ldw + h*64 + d]
// output slice Cst layout [32][8][32]
__device__ void gemmA_heads(const bf16_t* QH, const bf16_t* W, int ldw, int nBase, bf16_t* Cst) {
  int wave = threadIdx.x >> 5;
  for (int t = wave; t < 16; t += 8) {
    int h = t >> 1, nt = t & 1;
    v8f acc0 = vzero(), acc1 = vzero();
    const bf16_t* wp = W + (size_t)(nBase + nt * 16) * ldw + h * 64;
#pragma unroll
    for (int k0 = 0; k0 < 64; k0 += 32) {
      v16bf b = ldfragB(wp + k0, ldw);
      acc0 = wmma_bf16(ldfragA(QH + h * 64 + k0, HDIM), b, acc0);
      acc1 = wmma_bf16(ldfragA(QH + 16 * HDIM + h * 64 + k0, HDIM), b, acc1);
    }
    stfrag_bf16(Cst + h * 32 + nt * 16, 256, acc0, false);
    stfrag_bf16(Cst + 16 * 256 + h * 32 + nt * 16, 256, acc1, false);
  }
}

// ---------------- shared-state plumbing ----------------
struct Params {
  const float* x;
  const bf16_t *q1T, *q2T;
  const bf16_t *wq2dT, *wk2d, *wv2dT, *wo2dT, *wp2dT;
  const bf16_t *wq1dT, *wk1d, *wv1dT, *wo1dT;
  const bf16_t *ow1T, *ow2T, *rel;
  bf16_t* sa;
  float* out;
};
struct Smem {
  bf16_t* q;     // [32][512] current query / attn output
  bf16_t* t;     // [32][512] qh / pre-wo attention output
  bf16_t* mem;   // [32][8][32] gathered x slice (bf16)
  bf16_t* work;  // [32][8][32] A-tilde slice / ctx slice
  bf16_t* p;     // [32][64]   next-stage 7x7 prediction bias
  float*  logit; // [32][8][24]
};

__device__ void stage_mem(const float* xg, int bBase, int soff, int m0, int mc, int c0,
                          bf16_t* mem) {
  int tid = threadIdx.x;
  int tot = BT * mc * 32;
  for (int i = tid; i < tot; i += 256) {
    int cc = i & 31;
    int mi = (i >> 5) % mc;
    int b  = i / (mc * 32);
    int pos = POS[soff + m0 + mi];
    float v = xg[(size_t)(bBase + b) * 12544 + (size_t)(c0 + cc) * 49 + pos];
    mem[(b * 8 + mi) * 32 + cc] = (bf16_t)v;
  }
}

__device__ void softmax_rows(float* LG, const bf16_t* pb, int soff, int M) {
  int tid = threadIdx.x;  // exactly one (b,h) row per thread: 32*8 == 256
  int b = tid >> 3, h = tid & 7;
  float* L = LG + (b * 8 + h) * 24;
  float mx = -3.0e38f;
  for (int m = 0; m < M; ++m) {
    float bias = pb ? (float)pb[b * 64 + POS[soff + m]] : 0.f;
    float v = L[m] * 0.125f + bias;
    mx = fmaxf(mx, v);
  }
  float sum = 0.f;
  for (int m = 0; m < M; ++m) {
    float bias = pb ? (float)pb[b * 64 + POS[soff + m]] : 0.f;
    float e = __expf(L[m] * 0.125f + bias - mx);
    L[m] = e; sum += e;
  }
  float inv = 1.f / sum;
  for (int m = 0; m < M; ++m) L[m] *= inv;
}

// ---------------- attention stages ----------------
__device__ void attn2d_stage(int s, int bBase, const Params& P, const Smem& sm) {
  const int tid = threadIdx.x;
  const int wave = tid >> 5;
  const int M = SM[s], soff = SOFF[s];

  // qh = q @ wq^T   -> sm.t
  gemm16(sm.q, HDIM, P.wq2dT + (size_t)s * HDIM * HDIM, HDIM, sm.t, HDIM, HDIM, HDIM, false);
  __syncthreads();

  // logits init with relative-position term: sum_d qh[b,hd]*rel[pos,hd]
  for (int i = tid; i < BT * NHEAD * M; i += 256) {
    int m = i % M, h = (i / M) & 7, b = i / (M * 8);
    const bf16_t* qp = sm.t + b * HDIM + h * HD;
    const bf16_t* rp = P.rel + (size_t)POS[soff + m] * HDIM + h * HD;
    float acc = 0.f;
    for (int d = 0; d < HD; ++d) acc += (float)qp[d] * (float)rp[d];
    sm.logit[(b * 8 + h) * 24 + m] = acc;
  }
  __syncthreads();

  // logits pass: logits[b,h,m] += sum_c Atilde[b,h,c]*mem[b,m,c]  (c in chunks of 32)
  for (int c0 = 0; c0 < CCH; c0 += 32) {
    gemmA_heads(sm.t, P.wk2d + (size_t)s * CCH * HDIM, HDIM, c0, sm.work);
    __syncthreads();
    for (int m0 = 0; m0 < M; m0 += 8) {
      int mc = M - m0; if (mc > 8) mc = 8;
      stage_mem(P.x, bBase, soff, m0, mc, c0, sm.mem);
      __syncthreads();
      for (int i = tid; i < BT * NHEAD * mc; i += 256) {
        int mi = i % mc, h = (i / mc) & 7, b = i / (mc * 8);
        const bf16_t* ap = sm.work + (b * 8 + h) * 32;
        const bf16_t* mp = sm.mem + (b * 8 + mi) * 32;
        float acc = 0.f;
        for (int cc = 0; cc < 32; ++cc) acc += (float)ap[cc] * (float)mp[cc];
        sm.logit[(b * 8 + h) * 24 + m0 + mi] += acc;
      }
      __syncthreads();
    }
  }

  softmax_rows(sm.logit, s > 0 ? sm.p : (const bf16_t*)nullptr, soff, M);
  __syncthreads();

  // ctx pass: ctx_h[b,c] = sum_m w*mem; attn_pre_wo[b, h*64+d] = ctx_h @ wv_h (K accumulated over c-chunks)
  v8f accs[8];
#pragma unroll
  for (int j = 0; j < 8; ++j) accs[j] = vzero();
  for (int c0 = 0; c0 < CCH; c0 += 32) {
    float cacc[32];
#pragma unroll
    for (int j = 0; j < 32; ++j) cacc[j] = 0.f;
    for (int m0 = 0; m0 < M; m0 += 8) {
      int mc = M - m0; if (mc > 8) mc = 8;
      stage_mem(P.x, bBase, soff, m0, mc, c0, sm.mem);
      __syncthreads();
#pragma unroll
      for (int j = 0; j < 32; ++j) {
        int e = tid + j * 256;
        int cc = e & 31, h = (e >> 5) & 7, b = e >> 8;
        const float* wr = sm.logit + (b * 8 + h) * 24 + m0;
        float a = cacc[j];
        for (int mi = 0; mi < mc; ++mi)
          a += wr[mi] * (float)sm.mem[(b * 8 + mi) * 32 + cc];
        cacc[j] = a;
      }
      __syncthreads();
    }
#pragma unroll
    for (int j = 0; j < 32; ++j) sm.work[tid + j * 256] = (bf16_t)cacc[j];
    __syncthreads();
#pragma unroll
    for (int j = 0; j < 4; ++j) {
      int t = wave * 4 + j, h = t >> 2, nt = t & 3;
      v16bf bb = ldfragB(P.wv2dT + (size_t)s * HDIM * CCH + (size_t)(h * 64 + nt * 16) * CCH + c0, CCH);
      accs[2*j]   = wmma_bf16(ldfragA(sm.work + h * 32, 256), bb, accs[2*j]);
      accs[2*j+1] = wmma_bf16(ldfragA(sm.work + 16 * 256 + h * 32, 256), bb, accs[2*j+1]);
    }
    __syncthreads();
  }
#pragma unroll
  for (int j = 0; j < 4; ++j) {
    int t = wave * 4 + j, h = t >> 2, nt = t & 3;
    stfrag_bf16(sm.t + h * 64 + nt * 16, HDIM, accs[2*j], false);
    stfrag_bf16(sm.t + 16 * HDIM + h * 64 + nt * 16, HDIM, accs[2*j+1], false);
  }
  __syncthreads();

  // out = attn_pre_wo @ wo^T -> sm.q (becomes next query)
  gemm16(sm.t, HDIM, P.wo2dT + (size_t)s * HDIM * HDIM, HDIM, sm.q, HDIM, HDIM, HDIM, false);
  __syncthreads();
  if (s < 3) { // next-stage prediction bias p = out @ wp^T
    gemm16(sm.q, HDIM, P.wp2dT + (size_t)s * 64 * HDIM, HDIM, sm.p, 64, 64, HDIM, false);
    __syncthreads();
  }
}

__device__ void attn1d_stage(int t1, int nrows, int bBase, const Params& P, const Smem& sm) {
  const int tid = threadIdx.x;
  const int wave = tid >> 5;

  gemm16(sm.q, HDIM, P.wq1dT + (size_t)t1 * HDIM * HDIM, HDIM, sm.t, HDIM, HDIM, HDIM, false);
  __syncthreads();
  for (int i = tid; i < BT * NHEAD * 24; i += 256) sm.logit[i] = 0.f;
  __syncthreads();

  for (int e0 = 0; e0 < HDIM; e0 += 32) {
    gemmA_heads(sm.t, P.wk1d + (size_t)t1 * HDIM * HDIM, HDIM, e0, sm.work);
    __syncthreads();
    for (int i = tid; i < BT * NHEAD * nrows; i += 256) {
      int m = i % nrows, h = (i / nrows) & 7, b = i / (nrows * 8);
      const bf16_t* ap = sm.work + (b * 8 + h) * 32;
      const bf16_t* sp = P.sa + ((size_t)(bBase + b) * 4 + m) * HDIM + e0;
      float acc = 0.f;
      for (int ec = 0; ec < 32; ++ec) acc += (float)ap[ec] * (float)sp[ec];
      sm.logit[(b * 8 + h) * 24 + m] += acc;
    }
    __syncthreads();
  }

  softmax_rows(sm.logit, nullptr, 0, nrows);
  __syncthreads();

  v8f accs[8];
#pragma unroll
  for (int j = 0; j < 8; ++j) accs[j] = vzero();
  for (int e0 = 0; e0 < HDIM; e0 += 32) {
#pragma unroll
    for (int j = 0; j < 32; ++j) {
      int e = tid + j * 256;
      int cc = e & 31, h = (e >> 5) & 7, b = e >> 8;
      const float* wr = sm.logit + (b * 8 + h) * 24;
      float a = 0.f;
      for (int m = 0; m < nrows; ++m)
        a += wr[m] * (float)P.sa[((size_t)(bBase + b) * 4 + m) * HDIM + e0 + cc];
      sm.work[tid + j * 256] = (bf16_t)a;
    }
    __syncthreads();
#pragma unroll
    for (int j = 0; j < 4; ++j) {
      int t = wave * 4 + j, h = t >> 2, nt = t & 3;
      v16bf bb = ldfragB(P.wv1dT + (size_t)t1 * HDIM * HDIM + (size_t)(h * 64 + nt * 16) * HDIM + e0, HDIM);
      accs[2*j]   = wmma_bf16(ldfragA(sm.work + h * 32, 256), bb, accs[2*j]);
      accs[2*j+1] = wmma_bf16(ldfragA(sm.work + 16 * 256 + h * 32, 256), bb, accs[2*j+1]);
    }
    __syncthreads();
  }
#pragma unroll
  for (int j = 0; j < 4; ++j) {
    int t = wave * 4 + j, h = t >> 2, nt = t & 3;
    stfrag_bf16(sm.t + h * 64 + nt * 16, HDIM, accs[2*j], false);
    stfrag_bf16(sm.t + 16 * HDIM + h * 64 + nt * 16, HDIM, accs[2*j+1], false);
  }
  __syncthreads();
  gemm16(sm.t, HDIM, P.wo1dT + (size_t)t1 * HDIM * HDIM, HDIM, sm.q, HDIM, HDIM, HDIM, false);
  __syncthreads();
}

// ---------------- fused pipeline kernel ----------------
__global__ __launch_bounds__(256) void fused_attn_pipeline(Params P) {
  __shared__ __attribute__((aligned(16))) bf16_t s_q[BT * HDIM];
  __shared__ __attribute__((aligned(16))) bf16_t s_t[BT * HDIM];
  __shared__ __attribute__((aligned(16))) bf16_t s_mem[BT * 8 * 32];
  __shared__ __attribute__((aligned(16))) bf16_t s_work[BT * 8 * 32];
  __shared__ __attribute__((aligned(16))) float  s_logit[BT * 8 * 24];
  __shared__ __attribute__((aligned(16))) bf16_t s_p[BT * 64];
  Smem sm{s_q, s_t, s_mem, s_work, s_p, s_logit};

  const int tid = threadIdx.x;
  const int bBase = blockIdx.x * BT;

  if (tid == 0) { // warm L2 for weights (global_prefetch_b8)
    __builtin_prefetch(P.q1T, 0, 1);
    __builtin_prefetch(P.wq2dT, 0, 1);
    __builtin_prefetch(P.wk2d, 0, 1);
    __builtin_prefetch(P.wv2dT, 0, 1);
  }

  // q_init = relu(mem0 @ W1) @ W2, mem0 = x[:, :, 3, 3]
  for (int i = tid; i < BT * CCH; i += 256) {
    int b = i >> 8, c = i & 255;
    s_q[b * HDIM + c] = (bf16_t)P.x[(size_t)(bBase + b) * 12544 + (size_t)c * 49 + 24];
  }
  __syncthreads();
  gemm16(s_q, HDIM, P.q1T, CCH, s_t, HDIM, HDIM, CCH, true);
  __syncthreads();
  gemm16(s_t, HDIM, P.q2T, HDIM, s_q, HDIM, HDIM, HDIM, false);
  __syncthreads();

  for (int s = 0; s < 4; ++s) {
    attn2d_stage(s, bBase, P, sm);
    // append attn_s to sa cache (global, bf16)
    for (int i = tid; i < BT * HDIM; i += 256) {
      int b = i >> 9, j = i & 511;
      P.sa[((size_t)(bBase + b) * 4 + s) * HDIM + j] = s_q[b * HDIM + j];
    }
    __syncthreads();
    if (s > 0) attn1d_stage(s - 1, s + 1, bBase, P, sm);
  }

  // out = relu(attn_out @ ow1) @ ow2  -> f32 global
  gemm16(s_q, HDIM, P.ow1T, HDIM, s_t, HDIM, ODIM, HDIM, true);
  __syncthreads();
  gemm16_out(s_t, HDIM, P.ow2T, ODIM, P.out + (size_t)bBase * ODIM, ODIM, ODIM, ODIM);
}

// ---------------- weight prep kernels (f32 -> bf16, with transpose/pad) ----------------
__global__ void k_cvt(const float* src, bf16_t* dst, int n) {
  int i = blockIdx.x * 256 + threadIdx.x;
  if (i < n) dst[i] = (bf16_t)src[i];
}
__global__ void k_cvtT(const float* src, bf16_t* dst, int R, int C) { // src[R][C] -> dst[C][R]
  int i = blockIdx.x * 256 + threadIdx.x;
  if (i < R * C) {
    int r = i / C, c = i % C;
    dst[(size_t)c * R + r] = (bf16_t)src[i];
  }
}
__global__ void k_wpT(const float* src, bf16_t* dst) { // src[512][49] -> dst[64][512], zero-padded
  int i = blockIdx.x * 256 + threadIdx.x;
  if (i < 64 * 512) {
    int p = i >> 9, e = i & 511;
    dst[i] = (p < 49) ? (bf16_t)src[e * 49 + p] : (bf16_t)0.f;
  }
}
__global__ void k_rel(const float* rh, const float* rw, bf16_t* dst) { // [49][512] concat(rel_h[c],rel_w[c])
  int i = blockIdx.x * 256 + threadIdx.x;
  if (i < 49 * 512) {
    int p = i >> 9, j = i & 511;
    int col = p % 7;
    float v = (j < 256) ? rh[col * 256 + j] : rw[col * 256 + (j - 256)];
    dst[i] = (bf16_t)v;
  }
}

// ---------------- launch ----------------
extern "C" void kernel_launch(void* const* d_in, const int* in_sizes, int n_in,
                              void* d_out, int out_size, void* d_ws, size_t ws_size,
                              hipStream_t stream) {
  (void)in_sizes; (void)n_in; (void)out_size; (void)ws_size;
  const float* x      = (const float*)d_in[0];
  const float* rel_h  = (const float*)d_in[1];
  const float* rel_w  = (const float*)d_in[2];
  const float* qw1    = (const float*)d_in[3];
  const float* qw2    = (const float*)d_in[4];
  const float* a2d_wq = (const float*)d_in[5];
  const float* a2d_wk = (const float*)d_in[6];
  const float* a2d_wv = (const float*)d_in[7];
  const float* a2d_wo = (const float*)d_in[8];
  const float* a2d_wp = (const float*)d_in[9];
  const float* a1d_wq = (const float*)d_in[10];
  const float* a1d_wk = (const float*)d_in[11];
  const float* a1d_wv = (const float*)d_in[12];
  const float* a1d_wo = (const float*)d_in[13];
  const float* ow1    = (const float*)d_in[14];
  const float* ow2    = (const float*)d_in[15];

  bf16_t* w = (bf16_t*)d_ws;
  dim3 blk(256);
#define GRID(n) dim3(((n) + 255) / 256)

  k_cvtT<<<GRID(131072), blk, 0, stream>>>(qw1, w + OFF_Q1T, 256, 512);
  k_cvtT<<<GRID(262144), blk, 0, stream>>>(qw2, w + OFF_Q2T, 512, 512);
  for (int s = 0; s < 4; ++s) {
    k_cvtT<<<GRID(262144), blk, 0, stream>>>(a2d_wq + (size_t)s * 262144, w + OFF_WQ2DT + (size_t)s * 262144, 512, 512);
    k_cvt <<<GRID(131072), blk, 0, stream>>>(a2d_wk + (size_t)s * 131072, w + OFF_WK2D  + (size_t)s * 131072, 131072);
    k_cvtT<<<GRID(131072), blk, 0, stream>>>(a2d_wv + (size_t)s * 131072, w + OFF_WV2DT + (size_t)s * 131072, 256, 512);
    k_cvtT<<<GRID(262144), blk, 0, stream>>>(a2d_wo + (size_t)s * 262144, w + OFF_WO2DT + (size_t)s * 262144, 512, 512);
    k_wpT <<<GRID(32768),  blk, 0, stream>>>(a2d_wp + (size_t)s * 512 * 49, w + OFF_WP2DT + (size_t)s * 32768);
  }
  for (int t = 0; t < 3; ++t) {
    k_cvtT<<<GRID(262144), blk, 0, stream>>>(a1d_wq + (size_t)t * 262144, w + OFF_WQ1DT + (size_t)t * 262144, 512, 512);
    k_cvt <<<GRID(262144), blk, 0, stream>>>(a1d_wk + (size_t)t * 262144, w + OFF_WK1D  + (size_t)t * 262144, 262144);
    k_cvtT<<<GRID(262144), blk, 0, stream>>>(a1d_wv + (size_t)t * 262144, w + OFF_WV1DT + (size_t)t * 262144, 512, 512);
    k_cvtT<<<GRID(262144), blk, 0, stream>>>(a1d_wo + (size_t)t * 262144, w + OFF_WO1DT + (size_t)t * 262144, 512, 512);
  }
  k_cvtT<<<GRID(131072), blk, 0, stream>>>(ow1, w + OFF_OW1T, 512, 256);
  k_cvtT<<<GRID(65536),  blk, 0, stream>>>(ow2, w + OFF_OW2T, 256, 256);
  k_rel <<<GRID(25088),  blk, 0, stream>>>(rel_h, rel_w, w + OFF_REL);

  Params P;
  P.x = x;
  P.q1T   = w + OFF_Q1T;   P.q2T   = w + OFF_Q2T;
  P.wq2dT = w + OFF_WQ2DT; P.wk2d  = w + OFF_WK2D;
  P.wv2dT = w + OFF_WV2DT; P.wo2dT = w + OFF_WO2DT; P.wp2dT = w + OFF_WP2DT;
  P.wq1dT = w + OFF_WQ1DT; P.wk1d  = w + OFF_WK1D;
  P.wv1dT = w + OFF_WV1DT; P.wo1dT = w + OFF_WO1DT;
  P.ow1T  = w + OFF_OW1T;  P.ow2T  = w + OFF_OW2T;  P.rel = w + OFF_REL;
  P.sa    = w + OFF_SA;
  P.out   = (float*)d_out;

  fused_attn_pipeline<<<dim3(BATCH / BT), blk, 0, stream>>>(P);
#undef GRID
}
